// model4_27814208209095
// MI455X (gfx1250) — compile-verified, matrix operand order checked by reading, not run on hardware
//
#include <hip/hip_runtime.h>
#include <hip/hip_bf16.h>

// Problem constants (from reference): B,T,H,D
#define Bn 256
#define Tn 256
#define Hn 16
#define Dn 88
#define KSEG 96             // per-kind K segment (88 used + 8 zero pad)
#define KP 288              // 3 kinds * 96 = 9 chunks of K=32
#define NCH 9
#define ROWP 96             // padded LDS row pitch (halves)

typedef __attribute__((ext_vector_type(16))) _Float16 v16h;
typedef __attribute__((ext_vector_type(8)))  float    v8f;

// workspace layout (bytes)
#define COEF_BYTES (NCH * 32 * 16 * 2)           // 9216: f16 coef in B-fragment order
#define CONSTH_OFF COEF_BYTES                    // 9216: 16 floats
#define EMIT_OFF   (CONSTH_OFF + 64)             // 9280: B*T*H f32 = 4MB
#define RES_OFF    (EMIT_OFF + Bn * Tn * Hn * 4) // per-batch results (256 f32)

__device__ __forceinline__ int koff16(int j, int hi) {
  // 16-bit A/B operand K striping within a 32-wide chunk (ISA layout):
  // halves 0..7 -> K = j + 8*hi ; halves 8..15 -> K = 16 + (j-8) + 8*hi
  return (j < 8) ? (j + hi * 8) : (16 + (j - 8) + hi * 8);
}

// ---------------------------------------------------------------------------
// Kernel 1: build f16 coefficient matrix (pre-swizzled into per-lane WMMA
// B-fragment layout) + per-state emission constant.
// emit[b,t,h] = sum_d [ (y*c)(A1-A0) + y*A0 + c*(L1-L0) ] + sum_d L0
//   A_c = log(p_c) - log1p(-p_c),  L_c = log1p(-p_c),  p_c = probs_y[h,c,d]
// Logical feature index: kf = kind*96 + d (kind 0: y*c, 1: y, 2: c);
// d>=88 padded zero. Fragment entry e = (ch*32 + lane)*16 + j holds
// coef_logical[kf = ch*32 + koff16(j, lane>>4)][h = lane&15].
// ---------------------------------------------------------------------------
__global__ void hmm_prep(const float* __restrict__ py,
                         _Float16* __restrict__ coef,
                         float* __restrict__ constH) {
  for (int e = threadIdx.x; e < NCH * 32 * 16; e += blockDim.x) {
    const int j    = e & 15;
    const int lane = (e >> 4) & 31;
    const int ch   = e >> 9;
    const int hi   = lane >> 4;
    const int h    = lane & 15;
    const int kf   = ch * 32 + koff16(j, hi);
    const int kind = ch / 3;                 // chunks 0-2:y*c, 3-5:y, 6-8:c
    const int d    = kf - kind * KSEG;
    float v = 0.f;
    if (d < Dn) {
      float p0 = py[(h * 2 + 0) * Dn + d];
      float p1 = py[(h * 2 + 1) * Dn + d];
      float m0 = log1pf(-p0), m1 = log1pf(-p1);
      float A0 = __logf(p0) - m0, A1 = __logf(p1) - m1;
      v = (kind == 0) ? (A1 - A0) : (kind == 1) ? A0 : (m1 - m0);
    }
    coef[e] = (_Float16)v;
  }
  if (threadIdx.x < Hn) {
    int h = threadIdx.x;
    float s = 0.f;
    for (int d = 0; d < Dn; ++d) s += log1pf(-py[(h * 2) * Dn + d]);
    constH[h] = s;
  }
}

// ---------------------------------------------------------------------------
// Kernel 2: emission GEMM via v_wmma_f32_16x16x32_f16.
// Block = 8 waves = 8 row-tiles = 128 contiguous (b,t) rows (half a batch).
// LDS: fragment-ordered coef (9216 B) + 129 sequence rows as f16 (24768 B).
// B fragments: one 32B contiguous ds load per chunk per lane.
// A fragments: contiguous 16B runs per 8 halves (vectorizes to ds_load_b128),
// kind 0 is a packed-f16 product of the y and prev-y runs. No branches.
// ---------------------------------------------------------------------------
__global__ void __launch_bounds__(256) hmm_emit_wmma(
    const float* __restrict__ seq,
    const _Float16* __restrict__ coef,
    const float* __restrict__ constH,
    float* __restrict__ emit) {
  __shared__ _Float16 scoef[NCH * 32 * 16];  // 9216 B, B-fragment order
  __shared__ _Float16 sy[129 * ROWP];        // 24768 B

  const int tid  = threadIdx.x;
  const int wave = tid >> 5;
  const int lane = tid & 31;
  const int blockRow0 = blockIdx.x * 128;    // first (b,t) row of this block

  // stage coefficient fragments (dword-wide copies)
  for (int i = tid; i < NCH * 32 * 16 / 2; i += 256)
    ((unsigned int*)scoef)[i] = ((const unsigned int*)coef)[i];

  // stage sequence rows blockRow0-1 .. blockRow0+127 as f16 (row i = blockRow0-1+i)
  const bool firstIsPrevBatch = ((blockRow0 & (Tn - 1)) == 0);
  for (int i = wave; i < 129; i += 8) {
    if (i == 0 && firstIsPrevBatch) {
      for (int dd = lane; dd < Dn; dd += 32) sy[dd] = (_Float16)0.f;
    } else {
      const float* src = seq + (size_t)(blockRow0 - 1 + i) * Dn;
      for (int dd = lane; dd < Dn; dd += 32)
        sy[i * ROWP + dd] = (_Float16)src[dd];
    }
  }
  // zero the 8-half pad of every row
  for (int e = tid; e < 129 * 8; e += 256)
    sy[(e >> 3) * ROWP + Dn + (e & 7)] = (_Float16)0.f;
  __syncthreads();

  const int l  = lane & 15;                  // A row (M) / B,D column (N)
  const int hi = lane >> 4;                  // half-wave K-striping selector

  // B fragments: contiguous 32B per (chunk, lane) in LDS
  const v16h* bf = (const v16h*)scoef;

  // this wave's tile: rows row0..row0+15; LDS row index = local row + 1
  const int row0 = blockRow0 + wave * 16;
  const _Float16* yl = sy + (wave * 16 + l + 1) * ROWP;  // y[b,t]
  const _Float16* cl = yl - ROWP;                        // y[b,t-1] (or zeros)

  v8f acc = {};
#pragma unroll
  for (int ch = 0; ch < NCH; ++ch) {
    const int kind  = ch / 3;                // compile-time after unroll
    const int dbase = (ch - kind * 3) * 32;
    v16h a;
#pragma unroll
    for (int j = 0; j < 16; ++j) {
      const int d = dbase + koff16(j, hi);
      _Float16 v;
      if (kind == 0)      v = yl[d] * cl[d]; // y AND prev (binary product)
      else if (kind == 1) v = yl[d];
      else                v = cl[d];
      a[j] = v;
    }
    acc = __builtin_amdgcn_wmma_f32_16x16x32_f16(
        /*neg_a=*/false, a, /*neg_b=*/false, bf[ch * 32 + lane],
        /*c_mod=*/(short)0, acc, /*reuse_a=*/false, /*reuse_b=*/false);
  }

  // D layout: lanes 0-15 -> rows 0..7, lanes 16-31 -> rows 8..15; col N = l.
  const float cst = constH[l];
#pragma unroll
  for (int i = 0; i < 8; ++i) {
    const int m = i + 8 * hi;
    emit[(size_t)(row0 + m) * Hn + l] = acc[i] + cst;
  }
}

// ---------------------------------------------------------------------------
// Kernel 3: forward recursion. 16 lanes per batch (2 batches per wave32);
// alpha in registers, 16-lane all-gather via __shfl, prefetch emit rows ahead.
// ---------------------------------------------------------------------------
__global__ void __launch_bounds__(256) hmm_forward(
    const float* __restrict__ emit,
    const int* __restrict__ lengths,
    const float* __restrict__ px,
    float* __restrict__ res) {
  const int lane = threadIdx.x & 31;
  const int wave = threadIdx.x >> 5;
  const int sub  = lane >> 4;     // which batch within the wave
  const int h    = lane & 15;     // destination state h'
  const int b    = blockIdx.x * 16 + wave * 2 + sub;
  const int base = sub * 16;      // shuffle base lane for this batch

  float lt[Hn];                   // log_trans[:, h']
#pragma unroll
  for (int j = 0; j < Hn; ++j) lt[j] = __logf(px[j * Hn + h]);

  float alpha = (h == 0) ? 0.f : -1e9f;
  const int len = lengths[b];
  const float* em = emit + (size_t)b * Tn * Hn;

  for (int t = 0; t < Tn; ++t) {
    __builtin_prefetch(em + (t + 6) * Hn, 0, 0);  // global_prefetch_b8 ahead of chain
    float aj[Hn];
#pragma unroll
    for (int j = 0; j < Hn; ++j) aj[j] = __shfl(alpha, base + j, 32);
    float m_on = -3.0e38f, m_all = -3.0e38f;
#pragma unroll
    for (int j = 0; j < Hn; ++j) {
      m_all = fmaxf(m_all, aj[j]);
      m_on  = fmaxf(m_on,  aj[j] + lt[j]);
    }
    float s_on = 0.f, s_all = 0.f;
#pragma unroll
    for (int j = 0; j < Hn; ++j) {
      s_on  += __expf(aj[j] + lt[j] - m_on);
      s_all += __expf(aj[j] - m_all);
    }
    const float e     = em[t * Hn + h];
    const float a_on  = m_on  + __logf(s_on)  + e;
    const float a_off = m_all + __logf(s_all);
    alpha = (t < len) ? a_on : a_off;
  }

  // final logsumexp over states
  float av[Hn];
#pragma unroll
  for (int j = 0; j < Hn; ++j) av[j] = __shfl(alpha, base + j, 32);
  float m = -3.0e38f;
#pragma unroll
  for (int j = 0; j < Hn; ++j) m = fmaxf(m, av[j]);
  float s = 0.f;
#pragma unroll
  for (int j = 0; j < Hn; ++j) s += __expf(av[j] - m);
  if (h == 0) res[b] = m + __logf(s);
}

// ---------------------------------------------------------------------------
// Kernel 4: deterministic tree reduction over 256 per-batch results.
// ---------------------------------------------------------------------------
__global__ void hmm_reduce(const float* __restrict__ res, float* __restrict__ out) {
  __shared__ float s[Bn];
  const int tid = threadIdx.x;
  s[tid] = res[tid];
  __syncthreads();
  for (int w = Bn / 2; w > 0; w >>= 1) {
    if (tid < w) s[tid] += s[tid + w];
    __syncthreads();
  }
  if (tid == 0) out[0] = s[0];
}

extern "C" void kernel_launch(void* const* d_in, const int* in_sizes, int n_in,
                              void* d_out, int out_size, void* d_ws, size_t ws_size,
                              hipStream_t stream) {
  const float* seq  = (const float*)d_in[0];  // sequences (B,T,D) f32
  const int*   lens = (const int*)d_in[1];    // lengths (B,) i32
  const float* px   = (const float*)d_in[2];  // probs_x (H,H) f32
  const float* py   = (const float*)d_in[3];  // probs_y (H,2,D) f32
  float* out = (float*)d_out;
  char*  ws  = (char*)d_ws;

  _Float16* coef   = (_Float16*)ws;
  float*    constH = (float*)(ws + CONSTH_OFF);
  float*    emit   = (float*)(ws + EMIT_OFF);
  float*    res    = (float*)(ws + RES_OFF);

  hmm_prep<<<1, 256, 0, stream>>>(py, coef, constH);
  hmm_emit_wmma<<<(Bn * Tn) / 128, 256, 0, stream>>>(seq, coef, constH, emit);
  hmm_forward<<<Bn / 16, 256, 0, stream>>>(emit, lens, px, res);
  hmm_reduce<<<1, Bn, 0, stream>>>(res, out);
}
